// PCATemplateMap_15522011808027
// MI455X (gfx1250) — matrix operand order repaired and checked by reading, not compiled, not run on hardware
//
#include <hip/hip_runtime.h>
#include <hip/hip_bf16.h>

// ---------------------------------------------------------------------------
// PCA template map, MI455X (gfx1250).
//   Q[b,c,t] = sum_i x[b,c,i] * Bt[t,i]          (bf16 WMMA, HBM-bound)
//   D[b,t,s] = sum_c Qt*Qs - Sm_t*Sm_s/C         (tiny Gram)
//   v        = top eigenvector of D (10x10 Jacobi)
//   mask     = v @ Bt
// ---------------------------------------------------------------------------

#define HWSZ   361
#define NTPL   10
#define BATCH  64
#define CH     2048
#define KSTEPS 12        // 12 * 32 = 384 >= 361 (K zero-padded in LDS)
#define KPAD   384
#define QN     16        // template dim padded 10 -> 16 for WMMA
#define NC     128       // channels per block
#define ROWS   392       // LDS row stride in bf16: 384 + 8 (bank-conflict pad)

typedef __attribute__((ext_vector_type(16))) __bf16 v16bf;
typedef __attribute__((ext_vector_type(8)))  __bf16 v8bf;
typedef __attribute__((ext_vector_type(8)))  float  v8f;

static __device__ inline __bf16 f2bf(float f) {
  unsigned int u = __float_as_uint(f);
  u += 0x7FFFu + ((u >> 16) & 1u);            // round-to-nearest-even
  unsigned short h = (unsigned short)(u >> 16);
  return __builtin_bit_cast(__bf16, h);
}

// ---------------------------------------------------------------------------
// Stage 1: Q[b, c, t] (C x 16 per batch) = X[b] (C x HW) @ Bt^T (HW x 16)
// Block = 1 batch x 128 channels, 256 threads (8 waves, 1 ctile each).
// x slab staged to LDS (coalesced float4, f32->bf16), K zero-padded to 384,
// then 12 unguarded chained v_wmma_f32_16x16x32_bf16 per wave.
// ---------------------------------------------------------------------------
__global__ __launch_bounds__(256)
void pca_stage1_wmma(const float* __restrict__ x,
                     const float* __restrict__ tmpl,
                     float* __restrict__ Q) {
  __shared__ __align__(32) __bf16 bfr[KSTEPS * 32 * 16];   // 12 KB B frags
  __shared__ __align__(16) __bf16 axs[NC * ROWS];          // ~98 KB A tile

  const int b   = blockIdx.y;
  const int c0  = blockIdx.x * NC;
  const int tid = threadIdx.x;

  // --- B fragments for all 12 K-steps, in contiguous K order (A matches):
  // lane L -> column N = L&15; lanes 0-15 hold K 0..15, lanes 16-31 K 16..31.
  for (int i = tid; i < KSTEPS * 32; i += 256) {
    const int step  = i >> 5;
    const int lane  = i & 31;
    const int n     = lane & 15;
    const int kbase = step * 32 + ((lane < 16) ? 0 : 16);
#pragma unroll
    for (int j = 0; j < 16; ++j) {
      const int k = kbase + j;
      const float v = (n < NTPL && k < HWSZ) ? tmpl[n * HWSZ + k] : 0.0f;
      bfr[i * 16 + j] = f2bf(v);
    }
  }

  // --- Zero-pad K = 361..383 (disjoint from fill addresses; no extra sync).
  for (int i = tid; i < NC * 32; i += 256) {
    const int ch = i >> 5;
    const int k  = 352 + (i & 31);
    if (k >= HWSZ) axs[ch * ROWS + k] = f2bf(0.0f);
  }

  // --- Fill LDS A-tile: the 128-channel slab is contiguous in global memory
  // (row stride == HWSZ), so stream it as coalesced float4 loads.
  const float4* slab = (const float4*)(x + ((size_t)b * CH + c0) * HWSZ);
  const int nvec = NC * HWSZ / 4;                 // 11552 float4 per block
  for (int i = tid; i < nvec; i += 256) {
    const float4 v = slab[i];
    const int flat = i * 4;
#pragma unroll
    for (int e = 0; e < 4; ++e) {
      const int f  = flat + e;
      const int ch = f / HWSZ;                    // mul-shift, no branch
      const int k  = f - ch * HWSZ;
      axs[ch * ROWS + k] = f2bf((&v.x)[e]);
    }
  }
  __syncthreads();

  // --- WMMA: 8 waves, one 16(ch) x 16(tmpl) tile each, K = 12 x 32.
  const int lane = tid & 31;                      // wave32
  const int ct   = tid >> 5;
  const int mrow = lane & 15;
  const int koff = (lane < 16) ? 0 : 16;
  const int moff = (lane < 16) ? 0 : 8;

  const __bf16* arow = &axs[(ct * 16 + mrow) * ROWS];

  v8f acc = {0.f, 0.f, 0.f, 0.f, 0.f, 0.f, 0.f, 0.f};
#pragma unroll
  for (int s = 0; s < KSTEPS; ++s) {
    const int k0 = s * 32;
    // A (16x32 bf16): lane holds K {koff..koff+15}
    const v8bf lo = *(const v8bf*)&arow[k0 + koff];
    const v8bf hi = *(const v8bf*)&arow[k0 + koff + 8];
    const v16bf af = __builtin_shufflevector(
        lo, hi, 0, 1, 2, 3, 4, 5, 6, 7, 8, 9, 10, 11, 12, 13, 14, 15);
    const v16bf bf = *(const v16bf*)&bfr[(s * 32 + lane) * 16];

    acc = __builtin_amdgcn_wmma_f32_16x16x32_bf16(
        /*neg_a=*/false, af, /*neg_b=*/false, bf,
        /*c_mod=*/(short)0, acc, /*reuse_a=*/false, /*reuse_b=*/false);
  }

  // D layout: VGPR v, lane L -> row M = v + (L>=16 ? 8:0), col N = L&15
  float* qb = Q + ((size_t)b * CH + (size_t)(c0 + ct * 16)) * QN;
#pragma unroll
  for (int v = 0; v < 8; ++v) {
    qb[(size_t)(v + moff) * QN + mrow] = acc[v];
  }
}

// ---------------------------------------------------------------------------
// Stage 2: per batch, D[t,s] = sum_c Q[c,t]*Q[c,s] - (sum Q_t)(sum Q_s)/C
// ---------------------------------------------------------------------------
__global__ __launch_bounds__(256)
void pca_stage2_gram(const float* __restrict__ Q, float* __restrict__ Dm) {
  const int b = blockIdx.x;
  const int t = threadIdx.x >> 4;
  const int s = threadIdx.x & 15;
  const float* Qb = Q + (size_t)b * CH * QN;
  float g = 0.f, st = 0.f, ss = 0.f;
  for (int c = 0; c < CH; ++c) {
    const float qt = Qb[(size_t)c * QN + t];
    const float qs = Qb[(size_t)c * QN + s];
    g += qt * qs;
    st += qt;
    ss += qs;
  }
  Dm[b * 256 + t * 16 + s] = g - st * ss * (1.0f / (float)CH);
}

// ---------------------------------------------------------------------------
// Stage 3: 10x10 symmetric eigensolve (cyclic Jacobi), one matrix per thread.
// ---------------------------------------------------------------------------
__global__ void pca_stage3_eigh(const float* __restrict__ Dm,
                                float* __restrict__ vb,
                                float* __restrict__ out) {
  const int b = threadIdx.x;
  if (b >= BATCH) return;

  float a[NTPL][NTPL], V[NTPL][NTPL];
  for (int i = 0; i < NTPL; ++i)
    for (int j = 0; j < NTPL; ++j) {
      a[i][j] = Dm[b * 256 + i * 16 + j];
      V[i][j] = (i == j) ? 1.0f : 0.0f;
    }

  for (int sweep = 0; sweep < 12; ++sweep) {
    for (int p = 0; p < NTPL - 1; ++p) {
      for (int q = p + 1; q < NTPL; ++q) {
        const float apq = a[p][q];
        if (fabsf(apq) < 1e-12f) continue;
        const float theta = (a[q][q] - a[p][p]) / (2.0f * apq);
        const float tt = ((theta >= 0.f) ? 1.0f : -1.0f) /
                         (fabsf(theta) + sqrtf(theta * theta + 1.0f));
        const float cc = 1.0f / sqrtf(tt * tt + 1.0f);
        const float sn = tt * cc;
        for (int k = 0; k < NTPL; ++k) {        // A = A * J
          const float akp = a[k][p], akq = a[k][q];
          a[k][p] = cc * akp - sn * akq;
          a[k][q] = sn * akp + cc * akq;
        }
        for (int k = 0; k < NTPL; ++k) {        // A = J^T * A
          const float apk = a[p][k], aqk = a[q][k];
          a[p][k] = cc * apk - sn * aqk;
          a[q][k] = sn * apk + cc * aqk;
        }
        for (int k = 0; k < NTPL; ++k) {        // V = V * J
          const float vkp = V[k][p], vkq = V[k][q];
          V[k][p] = cc * vkp - sn * vkq;
          V[k][q] = sn * vkp + cc * vkq;
        }
      }
    }
  }

  int imax = 0;
  for (int i = 1; i < NTPL; ++i)
    if (a[i][i] > a[imax][imax]) imax = i;

  float v[NTPL];
  int jmax = 0;
  for (int k = 0; k < NTPL; ++k) {
    v[k] = V[k][imax];
    if (fabsf(v[k]) > fabsf(v[jmax])) jmax = k;
  }
  const float sgn = (v[jmax] < 0.f) ? -1.0f : 1.0f;
  for (int k = 0; k < NTPL; ++k) {
    const float vk = sgn * v[k];
    vb[b * 16 + k] = vk;
    out[BATCH * HWSZ + b * NTPL + k] = vk;     // v output (tail of d_out)
  }
}

// ---------------------------------------------------------------------------
// Stage 4: mask[b, hw] = sum_t v[b,t] * Bt[t, hw]
// ---------------------------------------------------------------------------
__global__ void pca_stage4_mask(const float* __restrict__ vb,
                                const float* __restrict__ tmpl,
                                float* __restrict__ out) {
  const int b  = blockIdx.x;
  const int hw = threadIdx.x;
  if (hw >= HWSZ) return;
  float m = 0.f;
#pragma unroll
  for (int t = 0; t < NTPL; ++t) m += vb[b * 16 + t] * tmpl[t * HWSZ + hw];
  out[(size_t)b * HWSZ + hw] = m;
}

// ---------------------------------------------------------------------------
extern "C" void kernel_launch(void* const* d_in, const int* in_sizes, int n_in,
                              void* d_out, int out_size, void* d_ws,
                              size_t ws_size, hipStream_t stream) {
  const float* x    = (const float*)d_in[0];   // (64, 2048, 19, 19) f32
  const float* tmpl = (const float*)d_in[1];   // (10, 19, 19) f32
  float* out = (float*)d_out;                  // mask (64*361) ++ v (64*10)

  float* Q  = (float*)d_ws;                    // 64*2048*16 f32  (8 MB)
  float* Dm = Q + (size_t)BATCH * CH * QN;     // 64*256 f32
  float* vb = Dm + BATCH * 256;                // 64*16 f32

  dim3 g1(CH / NC, BATCH);
  pca_stage1_wmma<<<g1, 256, 0, stream>>>(x, tmpl, Q);
  pca_stage2_gram<<<BATCH, 256, 0, stream>>>(Q, Dm);
  pca_stage3_eigh<<<1, 64, 0, stream>>>(Dm, vb, out);
  pca_stage4_mask<<<BATCH, 384, 0, stream>>>(vb, tmpl, out);
}
